// GuidedLigandContextWrapper_80616536146582
// MI455X (gfx1250) — matrix operand order repaired and checked by reading, not compiled, not run on hardware
//
#include <hip/hip_runtime.h>
#include <hip/hip_bf16.h>

typedef __attribute__((ext_vector_type(16))) _Float16 v16h;
typedef __attribute__((ext_vector_type(8)))  float    v8f;
typedef unsigned int u32x4 __attribute__((ext_vector_type(4)));
typedef int          i32x4 __attribute__((ext_vector_type(4)));
typedef int          i32x8 __attribute__((ext_vector_type(8)));

#define BB   32
#define LL   256
#define PP   4096
#define DD   128

#if defined(__has_builtin) && __has_builtin(__builtin_amdgcn_tensor_load_to_lds)
  #define HAVE_TDM 1
  #if __has_include(<hip/amd_detail/amd_gfx1250_TDM.h>)
    #define TDM_SIX_ARG 1   // therock/clang-23 toolchain: 6-arg builtin
  #else
    #define TDM_SIX_ARG 0   // ROCm 7.2 / clang-22: 5-arg builtin
  #endif
#else
  #define HAVE_TDM 0
#endif

__device__ __constant__ int AMAP[13] = {1, 6, 6, 7, 7, 8, 8, 9, 15, 16, 16, 17, 6};

#if HAVE_TDM
// 1-D TENSOR_LOAD_TO_LDS: copy `n` 4-byte elements from `gaddr` to LDS offset
// `lds_off`. D# packed per CDNA5 ISA §8.3/§8.4 (Group0/Group1 bit layout):
//   G0: count=1 | lds_addr | global_addr[56:0] | type=2
//   G1: data_size=2(4B); tensor_dim0=n; tensor_dim1=1; tile_dim0=n;
//       tile_dim1/2=0 (1-D); tensor_dim0_stride=n
// tile==tensor so no OOB handling can trigger.
__device__ __forceinline__ void tdm_load_row_f32(unsigned lds_off,
                                                 const void* gaddr,
                                                 unsigned n) {
  unsigned long long ga = (unsigned long long)gaddr;
  u32x4 g0;
  g0[0] = 1u;                                           // count=1, user mode
  g0[1] = lds_off;                                      // LDS byte address
  g0[2] = (unsigned)(ga & 0xffffffffull);               // global_addr[31:0]
  g0[3] = (unsigned)((ga >> 32) & 0x1ffffffull) | (2u << 30);  // [56:32]|type=2
  i32x8 g1;
  g1[0] = (int)(2u << 16);                              // data_size=4B
  g1[1] = (int)((n & 0xffffu) << 16);                   // tensor_dim0[15:0]
  g1[2] = (int)(((n >> 16) & 0xffffu) | (1u << 16));    // td0[31:16]|td1=1
  g1[3] = (int)((n & 0xffffu) << 16);                   // tile_dim0=n
  g1[4] = 0;                                            // tile_dim1/2 unused
  g1[5] = (int)n;                                       // td0_stride lo
  g1[6] = 0;
  g1[7] = 0;
  i32x4 z4 = {0, 0, 0, 0};
#if TDM_SIX_ARG
  i32x8 z8 = {0, 0, 0, 0, 0, 0, 0, 0};
  __builtin_amdgcn_tensor_load_to_lds(g0, g1, z4, z4, z8, 0);
#else
  __builtin_amdgcn_tensor_load_to_lds(g0, g1, z4, z4, 0);
#endif
}
#endif  // HAVE_TDM

// ---------------------------------------------------------------------------
// Kernel 1: embedding gathers + f32->f16 + TRANSPOSE into workspace tables.
//   hpocT[d][k]  : d in [0,128), k in [0,4096)      (batch-independent)
//   hligT[b][d][k]: k in [0,256)
//   woutT[n][k]  : n,k in [0,128)   (woutT[n][k] = w_out[k][n])
// Transposed so a WMMA B-tile column is 16 contiguous halfs (32B load).
// ---------------------------------------------------------------------------
__global__ void embed_transpose_kernel(const int* __restrict__ ligand_v,
                                       const int* __restrict__ pocket_z,
                                       const float* __restrict__ emb,
                                       const float* __restrict__ w_out,
                                       _Float16* __restrict__ hpocT,
                                       _Float16* __restrict__ hligT,
                                       _Float16* __restrict__ woutT) {
  int row = blockIdx.x;
  int d   = threadIdx.x;
  if (row < PP) {
    int z = pocket_z[row];
    hpocT[(size_t)d * PP + row] = (_Float16)emb[(size_t)z * DD + d];
  } else if (row < PP + BB * LL) {
    int i = row - PP;
    int b = i >> 8;        // i / LL
    int k = i & (LL - 1);  // i % LL
    int z = AMAP[ligand_v[i]];
    hligT[(size_t)b * DD * LL + (size_t)d * LL + k] =
        (_Float16)emb[(size_t)z * DD + d];
  } else {
    int r = row - PP - BB * LL;  // w_out row index (k)
    woutT[(size_t)d * DD + r] = (_Float16)w_out[(size_t)r * DD + d];
  }
}

// ---------------------------------------------------------------------------
// Kernel 2: fused RBF message passing + output GEMM + tanh + reduce.
// Grid = B * 4 quadrants. WG = 128 threads = 4 waves; wave w owns rows
// [quad*64 + 16w, +16) and ALL 128 columns as 8 f32 accumulators.
// A (16x32 RBF weights) built in registers in the 16-bit WMMA A layout;
// B tiles are direct 32-byte loads from the transposed f16 tables.
// Per-slab positions are DMA'd to LDS by the Tensor Data Mover (wave 0).
// ---------------------------------------------------------------------------
__global__ void __launch_bounds__(128)
affinity_kernel(const float* __restrict__ ligand_pos,
                const float* __restrict__ protein_pos,
                const _Float16* __restrict__ hpocT,
                const _Float16* __restrict__ hligT,
                const _Float16* __restrict__ woutT,
                float* __restrict__ partials) {
  __shared__ float lds_pos[32 * 3];        // slab atom positions (shared)
  __shared__ float lds_agg[4 * 16 * DD];   // per-wave agg blocks, 32 KB
  __shared__ float lds_red[128];

  const int tid  = threadIdx.x;
  const int lane = tid & 31;
  const int wave = tid >> 5;
  const int b    = blockIdx.x >> 2;
  const int m0   = (blockIdx.x & 3) << 6;  // 64-row quadrant base
  const int m    = lane & 15;              // A row in tile / B,C column in tile
  const int g    = lane >> 4;              // K sub-group selector
  const int row  = m0 + wave * 16 + m;     // this lane's global ligand row

  const float* lp = ligand_pos + ((size_t)b * LL + row) * 3;
  const float lx = lp[0], ly = lp[1], lz = lp[2];

  v8f c[8] = {};  // 16 rows x 128 cols, f32

  for (int phase = 0; phase < 2; ++phase) {
    const int       kdim    = phase ? LL : PP;
    const float*    posbase = phase ? (ligand_pos + (size_t)b * LL * 3)
                                    : (protein_pos + (size_t)b * PP * 3);
    const _Float16* bT      = phase ? (hligT + (size_t)b * DD * LL) : hpocT;
    const float     R2      = phase ? 25.0f : 36.0f;
    const float     invR2   = phase ? (1.0f / 25.0f) : (1.0f / 36.0f);

    for (int kbase = 0; kbase < kdim; kbase += 32) {
      __syncthreads();  // WAR on lds_pos: previous slab readers done
#if HAVE_TDM
      if (wave == 0) {
        tdm_load_row_f32((unsigned)(unsigned long long)(void*)lds_pos,
                         posbase + (size_t)kbase * 3, 96u);
#if defined(__has_builtin) && __has_builtin(__builtin_amdgcn_s_wait_tensorcnt)
        __builtin_amdgcn_s_wait_tensorcnt(0);
#else
        asm volatile("s_wait_tensorcnt 0x0" ::: "memory");
#endif
      }
#else
      if (tid < 96) lds_pos[tid] = posbase[(size_t)kbase * 3 + tid];
#endif
      __syncthreads();  // publish positions to all waves

      // ---- build A tile (16x32 RBF weights) in WMMA 16-bit A layout ----
      v16h a;
#pragma unroll
      for (int h = 0; h < 16; ++h) {
        const int koff = h + (g << 4);  // K within slab
        const float px = lds_pos[koff * 3 + 0];
        const float py = lds_pos[koff * 3 + 1];
        const float pz = lds_pos[koff * 3 + 2];
        const float dx = lx - px, dy = ly - py, dz = lz - pz;
        const float d2 = dx * dx + dy * dy + dz * dz;
        const bool excl = (phase != 0) && ((kbase + koff) == row);
        const float w = (d2 <= R2 && !excl)
                            ? __expf(-fmaxf(d2, 1e-12f) * invR2)
                            : 0.0f;
        a[h] = (_Float16)w;
      }

      // ---- 8 N-tiles: direct 32B B loads from transposed table + WMMA ----
#pragma unroll
      for (int t = 0; t < 8; ++t) {
        const int n = t * 16 + m;  // global column
        v16h bv = *(const v16h*)(bT + (size_t)n * kdim + kbase + (g << 4));
        c[t] = __builtin_amdgcn_wmma_f32_16x16x32_f16(false, a, false, bv,
                                                      (short)0, c[t], false,
                                                      false);
      }
    }
  }

  // ---- spill agg (16x128 f32) to this wave's private LDS region ----
  float* agg = lds_agg + wave * 16 * DD;
#pragma unroll
  for (int t = 0; t < 8; ++t)
#pragma unroll
    for (int r = 0; r < 8; ++r)
      agg[(r + (g << 3)) * DD + t * 16 + m] = c[t][r];
  // same-wave producer/consumer: no barrier needed (dscnt waits suffice)

  // ---- fused second GEMM: out = tanh(agg @ w_out), K = 128 (4 slabs) ----
  v8f o[8] = {};
#pragma unroll
  for (int slab = 0; slab < 4; ++slab) {
    v16h a2;
#pragma unroll
    for (int h = 0; h < 16; ++h) {
      const int koff = h + (g << 4);
      a2[h] = (_Float16)agg[m * DD + slab * 32 + koff];
    }
#pragma unroll
    for (int t = 0; t < 8; ++t) {
      const int n = t * 16 + m;
      v16h wv = *(const v16h*)(woutT + (size_t)n * DD + slab * 32 + (g << 4));
      o[t] = __builtin_amdgcn_wmma_f32_16x16x32_f16(false, a2, false, wv,
                                                    (short)0, o[t], false,
                                                    false);
    }
  }

  // ---- tanh + fixed-order block reduction -> one partial per WG ----
  float s = 0.0f;
#pragma unroll
  for (int t = 0; t < 8; ++t)
#pragma unroll
    for (int r = 0; r < 8; ++r) s += tanhf(o[t][r]);
  lds_red[tid] = s;
  __syncthreads();
  if (tid == 0) {
    float tt = 0.0f;
    for (int i = 0; i < 128; ++i) tt += lds_red[i];
    partials[blockIdx.x] = tt;
  }
}

// ---------------------------------------------------------------------------
// Kernel 3: deterministic per-batch reduction, negate + mean over L*D.
// ---------------------------------------------------------------------------
__global__ void finalize_kernel(const float* __restrict__ partials,
                                float* __restrict__ out) {
  int b = threadIdx.x;
  if (b < BB) {
    float s = 0.0f;
    for (int i = 0; i < 4; ++i) s += partials[b * 4 + i];
    out[b] = -s * (1.0f / (float)(LL * DD));
  }
}

extern "C" void kernel_launch(void* const* d_in, const int* in_sizes, int n_in,
                              void* d_out, int out_size, void* d_ws, size_t ws_size,
                              hipStream_t stream) {
  const float* ligand_pos  = (const float*)d_in[0];
  const int*   ligand_v    = (const int*)d_in[1];
  // d_in[2] batch_ligand, d_in[3] batch_protein: implied by sorted layout
  const float* protein_pos = (const float*)d_in[4];
  const int*   pocket_z    = (const int*)d_in[5];
  const float* emb_table   = (const float*)d_in[6];
  const float* w_out       = (const float*)d_in[7];

  // workspace layout (halfs, then floats): hpocT | hligT | woutT | partials
  _Float16* hpocT = (_Float16*)d_ws;
  _Float16* hligT = hpocT + (size_t)DD * PP;            // 524288 halfs
  _Float16* woutT = hligT + (size_t)BB * DD * LL;       // +1048576 halfs
  float*    parts = (float*)(woutT + (size_t)DD * DD);  // +16384 halfs

  embed_transpose_kernel<<<PP + BB * LL + DD, DD, 0, stream>>>(
      ligand_v, pocket_z, emb_table, w_out, hpocT, hligT, woutT);

  affinity_kernel<<<BB * 4, 128, 0, stream>>>(
      ligand_pos, protein_pos, hpocT, hligT, woutT, parts);

  finalize_kernel<<<1, 32, 0, stream>>>(parts, (float*)d_out);
}
